// GCN_LSTMRegressor_60524679135515
// MI455X (gfx1250) — compile-verified, matrix operand order checked by reading, not run on hardware
//
#include <hip/hip_runtime.h>
#include <hip/hip_bf16.h>

// ---- problem constants (match reference) ----
#define N_NODES 400
#define N_EDGES 6400
#define BATCH   32
#define TSTEPS  32
#define HFEAT   32
#define HID     256
#define G4      1024                 // 4*HID
#define K0      (N_NODES * HFEAT)    // 12800
#define MROWS   (BATCH * TSTEPS)     // 1024
#define OUTF    2

#define KCHUNK  64                   // K-tile staged in LDS per iteration
#define TILEB   (64 * KCHUNK * 2)    // bytes per (64 x KCHUNK) bf16 tile

typedef __attribute__((ext_vector_type(16))) __bf16 v16bf;
typedef __attribute__((ext_vector_type(8)))  float  v8f;

union Frag {
  v16bf v;
  uint4 q[2];
};

__device__ __forceinline__ unsigned short f2bf(float f) {
  unsigned int u = __float_as_uint(f);
  u += 0x7FFFu + ((u >> 16) & 1u);   // round-to-nearest-even
  return (unsigned short)(u >> 16);
}

__device__ __forceinline__ float sigmf(float x) {
  return 1.0f / (1.0f + __expf(-x));
}

__device__ __forceinline__ v8f wmma_bf16(const Frag& a, const Frag& b, v8f c) {
  return __builtin_amdgcn_wmma_f32_16x16x32_bf16(
      /*neg_a=*/false, a.v, /*neg_b=*/false, b.v,
      /*c_mod=*/(short)0, c, /*reuse_a=*/false, /*reuse_b=*/false);
}

// async 16B global -> LDS copy (per lane), tracked by ASYNCcnt
__device__ __forceinline__ void async_cp16(unsigned lds_off, const unsigned short* gptr) {
  unsigned long long ga = (unsigned long long)(size_t)gptr;
  asm volatile("global_load_async_to_lds_b128 %0, %1, off"
               :: "v"(lds_off), "v"(ga) : "memory");
}

// ---------------------------------------------------------------------------
// small utility kernels
// ---------------------------------------------------------------------------
__global__ void zero_kernel(float* __restrict__ p, int n) {
  int i = blockIdx.x * blockDim.x + threadIdx.x;
  if (i < n) p[i] = 0.0f;
}

__global__ void f2bf_kernel(const float* __restrict__ in,
                            unsigned short* __restrict__ out, int n) {
  int i = blockIdx.x * blockDim.x + threadIdx.x;
  if (i < n) out[i] = f2bf(in[i]);
}

__global__ void bias_combine(const float* __restrict__ a,
                             const float* __restrict__ b,
                             float* __restrict__ o, int n) {
  int i = blockIdx.x * blockDim.x + threadIdx.x;
  if (i < n) o[i] = a[i] + b[i];
}

// ---------------------------------------------------------------------------
// GraphConv pieces
// ---------------------------------------------------------------------------
__global__ void degree_kernel(const int* __restrict__ src,
                              const int* __restrict__ dst,
                              float* __restrict__ degO,
                              float* __restrict__ degI, int e) {
  int i = blockIdx.x * blockDim.x + threadIdx.x;
  if (i < e) {
    atomicAdd(degO + src[i], 1.0f);
    atomicAdd(degI + dst[i], 1.0f);
  }
}

__global__ void norm_kernel(float* __restrict__ degO,
                            float* __restrict__ degI, int n) {
  int i = blockIdx.x * blockDim.x + threadIdx.x;
  if (i < n) {
    float a = degO[i]; degO[i] = rsqrtf(a < 1.0f ? 1.0f : a);
    float c = degI[i]; degI[i] = rsqrtf(c < 1.0f ? 1.0f : c);
  }
}

// one block per edge; B*T = 1024 elements per edge
__global__ void edge_scatter(const float* __restrict__ in_feat,
                             const int* __restrict__ src,
                             const int* __restrict__ dst,
                             const float* __restrict__ ew,
                             const float* __restrict__ norm_src,
                             float* __restrict__ agg) {
  int e = blockIdx.x;
  int s = src[e], d = dst[e];
  float w = ew[e] * norm_src[s];
  const float* fp = in_feat + (size_t)s * (BATCH * TSTEPS);
  float* op = agg + (size_t)d * (BATCH * TSTEPS);
  for (int i = threadIdx.x; i < BATCH * TSTEPS; i += blockDim.x)
    atomicAdd(op + i, fp[i] * w);
}

// fuse: h = relu(agg * norm_dst * w1[f] + b1[f]); layout [B*T, N*HF] in bf16
__global__ void build_x0(const float* __restrict__ agg,
                         const float* __restrict__ norm_dst,
                         const float* __restrict__ w1,
                         const float* __restrict__ b1,
                         unsigned short* __restrict__ X0) {
  int idx = blockIdx.x * blockDim.x + threadIdx.x;
  if (idx >= MROWS * K0) return;
  int kk = idx % K0;           // n*HF + f
  int m  = idx / K0;           // b*T + t
  int n = kk >> 5;             // HF == 32
  int f = kk & 31;
  int b = m >> 5;              // T == 32
  int t = m & 31;
  float v = agg[(size_t)n * (BATCH * TSTEPS) + b * TSTEPS + t] * norm_dst[n] * w1[f] + b1[f];
  X0[idx] = f2bf(v > 0.0f ? v : 0.0f);
}

// ---------------------------------------------------------------------------
// WMMA GEMM with double-buffered async LDS staging:
//   C[M, Ng] = A[M, K](bf16) * B[Ng, K](bf16)^T + bias[Ng]
// grid = (M/64, Ng/64), block = 256 threads (8 waves).
// Per K-chunk (64): each thread issues 4 global_load_async_to_lds_b128 (2 for
// the 64x64 A tile, 2 for the 64x64 B tile) into the next buffer, waits
// s_wait_asynccnt for the current buffer (in-order completion), barriers, then
// feeds WMMA fragments from LDS (ds_load_b128).
// wave>>1 picks the 16-row M-subtile, wave&1 picks a 32-col N-half (2 tiles).
// ---------------------------------------------------------------------------
__global__ __launch_bounds__(256)
void wmma_gemm_abT(const unsigned short* __restrict__ A,
                   const unsigned short* __restrict__ Bm,
                   const float* __restrict__ bias,
                   float* __restrict__ C,
                   int M, int Ng, int K) {
  __shared__ __align__(16) unsigned short atile[2][64 * KCHUNK];
  __shared__ __align__(16) unsigned short btile[2][64 * KCHUNK];
  (void)M;

  int tid  = threadIdx.x;
  int lane = tid & 31;
  int wave = tid >> 5;
  int half = lane >> 4;
  int l15  = lane & 15;
  int kb   = half * 8;

  int mBase = blockIdx.x * 64;
  int nBase = blockIdx.y * 64;

  // loader mapping: 512 16B segments per tile, 2 per thread
  int ar0 = tid >> 3;               // rows 0..31
  int ac0 = (tid & 7) * 8;          // col offset in elements
  int ar1 = ar0 + 32;               // rows 32..63
  int ac1 = ac0;

  const unsigned short* aP0 = A  + (size_t)(mBase + ar0) * K + ac0;
  const unsigned short* aP1 = A  + (size_t)(mBase + ar1) * K + ac1;
  const unsigned short* bP0 = Bm + (size_t)(nBase + ar0) * K + ac0;
  const unsigned short* bP1 = Bm + (size_t)(nBase + ar1) * K + ac1;

  unsigned aL0 = (unsigned)(size_t)&atile[0][ar0 * KCHUNK + ac0];
  unsigned aL1 = (unsigned)(size_t)&atile[0][ar1 * KCHUNK + ac1];
  unsigned bL0 = (unsigned)(size_t)&btile[0][ar0 * KCHUNK + ac0];
  unsigned bL1 = (unsigned)(size_t)&btile[0][ar1 * KCHUNK + ac1];

  int nk = K / KCHUNK;

  // prologue: fill buffer 0
  async_cp16(aL0, aP0);
  async_cp16(aL1, aP1);
  async_cp16(bL0, bP0);
  async_cp16(bL1, bP1);

  v8f acc0 = {};
  v8f acc1 = {};
  int mtl  = (wave >> 1) * 16;      // local M row base
  int ntl0 = (wave & 1) * 32;       // local N base (two 16-col tiles)

  for (int i = 0; i < nk; i++) {
    int cur = i & 1;
    if (i + 1 < nk) {
      int nxt = (i + 1) & 1;
      int k0  = (i + 1) * KCHUNK;
      unsigned boff = (unsigned)nxt * TILEB;
      async_cp16(aL0 + boff, aP0 + k0);
      async_cp16(aL1 + boff, aP1 + k0);
      async_cp16(bL0 + boff, bP0 + k0);
      async_cp16(bL1 + boff, bP1 + k0);
      asm volatile("s_wait_asynccnt 0x4" ::: "memory");  // chunk i landed
    } else {
      asm volatile("s_wait_asynccnt 0x0" ::: "memory");
    }
    __syncthreads();                 // all waves' chunk-i copies visible

    const unsigned short* at = &atile[cur][0];
    const unsigned short* bt = &btile[cur][0];
    for (int ks = 0; ks < KCHUNK; ks += 32) {
      Frag a, b0, b1;
      const unsigned short* ap  = at + (mtl + l15) * KCHUNK + ks + kb;
      const unsigned short* bp0 = bt + (ntl0 + l15) * KCHUNK + ks + kb;
      const unsigned short* bp1 = bt + (ntl0 + 16 + l15) * KCHUNK + ks + kb;
      a.q[0]  = *(const uint4*)ap;
      a.q[1]  = *(const uint4*)(ap + 16);
      b0.q[0] = *(const uint4*)bp0;
      b0.q[1] = *(const uint4*)(bp0 + 16);
      b1.q[0] = *(const uint4*)bp1;
      b1.q[1] = *(const uint4*)(bp1 + 16);
      acc0 = wmma_bf16(a, b0, acc0);
      acc1 = wmma_bf16(a, b1, acc1);
    }
    __syncthreads();                 // reads done before buffer is re-filled
  }

  int mrow = half * 8;
  for (int r = 0; r < 8; r++) {
    int m  = mBase + mtl + mrow + r;
    int g0 = nBase + ntl0 + l15;
    int g1 = g0 + 16;
    C[(size_t)m * Ng + g0] = acc0[r] + bias[g0];
    C[(size_t)m * Ng + g1] = acc1[r] + bias[g1];
  }
}

// ---------------------------------------------------------------------------
// persistent single-block LSTM layer.
//   xW  : [B*T, 4H] fp32, row m = b*T + t, biases already folded in
//   Whh : [4H, H] bf16 (gate order i,f,g,o in row blocks of H)
//   hseq: [B*T, H] bf16 output (m = b*T + t)   -> feeds next layer's GEMM
//   hlast: [B, H] fp32, h at t = T-1           -> feeds FC
// h lives in LDS (bf16), c lives in registers per wave-owned tile.
// 8 waves * 4 combos cover the 32x1024 gate matrix; per combo four WMMA
// accumulators (i,f,g,o) are element-aligned so the gate nonlinearity is
// applied fully in-register.
// ---------------------------------------------------------------------------
__global__ __launch_bounds__(256)
void lstm_layer(const float* __restrict__ xW,
                const unsigned short* __restrict__ Whh,
                unsigned short* __restrict__ hseq,
                float* __restrict__ hlast) {
  __shared__ __align__(16) unsigned short hbuf[BATCH * HID];

  int lane = threadIdx.x & 31;
  int wave = threadIdx.x >> 5;
  int half = lane >> 4;
  int l15  = lane & 15;
  int kb   = half * 8;
  const size_t gs = (size_t)HID * HID;   // gate stride in Whh elements

  for (int i = threadIdx.x; i < BATCH * HID; i += 256) hbuf[i] = 0;

  float cst[4][8];
  for (int j = 0; j < 4; j++)
    for (int r = 0; r < 8; r++) cst[j][r] = 0.0f;

  __syncthreads();

  for (int t = 0; t < TSTEPS; t++) {
    float hv[4][8];
    for (int j = 0; j < 4; j++) {
      int cb = wave * 4 + j;
      int rt = cb >> 4;          // batch row-tile: 0..1
      int ht = cb & 15;          // hidden col-tile: 0..15
      int b0 = rt * 16 + half * 8;
      int hc = ht * 16 + l15;

      v8f gi, gf, gg, go;
      for (int r = 0; r < 8; r++) {
        const float* xp = xW + ((size_t)(b0 + r) * TSTEPS + t) * G4 + hc;
        gi[r] = xp[0];
        gf[r] = xp[HID];
        gg[r] = xp[2 * HID];
        go[r] = xp[3 * HID];
      }

      int arow = rt * 16 + l15;
      int wrow = ht * 16 + l15;
      for (int k0 = 0; k0 < HID; k0 += 32) {
        Frag a, bi, bf2, bg, bo;
        const unsigned short* ap = hbuf + arow * HID + k0 + kb;
        a.q[0] = *(const uint4*)ap;
        a.q[1] = *(const uint4*)(ap + 16);
        const unsigned short* wp = Whh + (size_t)wrow * HID + k0 + kb;
        bi.q[0]  = *(const uint4*)(wp);
        bi.q[1]  = *(const uint4*)(wp + 16);
        bf2.q[0] = *(const uint4*)(wp + gs);
        bf2.q[1] = *(const uint4*)(wp + gs + 16);
        bg.q[0]  = *(const uint4*)(wp + 2 * gs);
        bg.q[1]  = *(const uint4*)(wp + 2 * gs + 16);
        bo.q[0]  = *(const uint4*)(wp + 3 * gs);
        bo.q[1]  = *(const uint4*)(wp + 3 * gs + 16);
        gi = wmma_bf16(a, bi,  gi);
        gf = wmma_bf16(a, bf2, gf);
        gg = wmma_bf16(a, bg,  gg);
        go = wmma_bf16(a, bo,  go);
      }

      for (int r = 0; r < 8; r++) {
        float cv = sigmf(gf[r]) * cst[j][r] + sigmf(gi[r]) * tanhf(gg[r]);
        cst[j][r] = cv;
        hv[j][r] = sigmf(go[r]) * tanhf(cv);
      }
    }

    __syncthreads();   // everyone is done reading old h

    for (int j = 0; j < 4; j++) {
      int cb = wave * 4 + j;
      int rt = cb >> 4;
      int ht = cb & 15;
      int b0 = rt * 16 + half * 8;
      int hc = ht * 16 + l15;
      for (int r = 0; r < 8; r++) {
        unsigned short hb = f2bf(hv[j][r]);
        hbuf[(b0 + r) * HID + hc] = hb;
        hseq[((size_t)(b0 + r) * TSTEPS + t) * HID + hc] = hb;
        if (t == TSTEPS - 1) hlast[(b0 + r) * HID + hc] = hv[j][r];
      }
    }

    __syncthreads();   // new h visible before next step
  }
}

// ---------------------------------------------------------------------------
// FC on last timestep + output transpose to [N, B, OUT]
// ---------------------------------------------------------------------------
__global__ void fc_kernel(const float* __restrict__ h,
                          const float* __restrict__ Wfc,
                          const float* __restrict__ bfc,
                          float* __restrict__ out) {
  int idx = blockIdx.x * blockDim.x + threadIdx.x;
  if (idx >= BATCH * N_NODES * OUTF) return;
  int b = idx / (N_NODES * OUTF);
  int j = idx % (N_NODES * OUTF);
  const float* hr = h + (size_t)b * HID;
  const float* wr = Wfc + (size_t)j * HID;
  float s = bfc[j];
  for (int k = 0; k < HID; k++) s += hr[k] * wr[k];
  int n = j / OUTF, o = j % OUTF;
  out[((size_t)n * BATCH + b) * OUTF + o] = s;
}

// ---------------------------------------------------------------------------
extern "C" void kernel_launch(void* const* d_in, const int* in_sizes, int n_in,
                              void* d_out, int out_size, void* d_ws, size_t ws_size,
                              hipStream_t stream) {
  (void)in_sizes; (void)n_in; (void)out_size; (void)ws_size;

  const float* in_feat = (const float*)d_in[0];
  const int*   src     = (const int*)d_in[1];
  const int*   dst     = (const int*)d_in[2];
  const float* ew      = (const float*)d_in[3];
  const float* w1      = (const float*)d_in[4];
  const float* b1      = (const float*)d_in[5];
  const float* Wih0    = (const float*)d_in[6];
  const float* Whh0    = (const float*)d_in[7];
  const float* bih0    = (const float*)d_in[8];
  const float* bhh0    = (const float*)d_in[9];
  const float* Wih1    = (const float*)d_in[10];
  const float* Whh1    = (const float*)d_in[11];
  const float* bih1    = (const float*)d_in[12];
  const float* bhh1    = (const float*)d_in[13];
  const float* Wfc     = (const float*)d_in[14];
  const float* bfc     = (const float*)d_in[15];
  float* out = (float*)d_out;

  // ---- workspace carve-up (256B aligned blocks) ----
  char* ws = (char*)d_ws;
  size_t off = 0;
  auto carve = [&](size_t bytes) -> char* {
    char* p = ws + off;
    off = (off + bytes + 255) & ~(size_t)255;
    return p;
  };

  const int NBT = N_NODES * BATCH * TSTEPS;      // 409600
  // agg + degO + degI contiguous so one zero pass covers all three
  float* agg  = (float*)carve((size_t)(NBT + 2 * N_NODES) * sizeof(float));
  float* degO = agg + NBT;
  float* degI = degO + N_NODES;

  unsigned short* X0    = (unsigned short*)carve((size_t)MROWS * K0 * 2);
  unsigned short* W0b   = (unsigned short*)carve((size_t)G4 * K0 * 2);
  unsigned short* Whh0b = (unsigned short*)carve((size_t)G4 * HID * 2);
  unsigned short* Wih1b = (unsigned short*)carve((size_t)G4 * HID * 2);
  unsigned short* Whh1b = (unsigned short*)carve((size_t)G4 * HID * 2);
  float* bias0 = (float*)carve(G4 * sizeof(float));
  float* bias1 = (float*)carve(G4 * sizeof(float));
  float* xW    = (float*)carve((size_t)MROWS * G4 * sizeof(float));
  unsigned short* hseq0 = (unsigned short*)carve((size_t)MROWS * HID * 2);
  unsigned short* hseq1 = (unsigned short*)carve((size_t)MROWS * HID * 2);
  float* hlast = (float*)carve((size_t)BATCH * HID * sizeof(float));

  // ---- GraphConv ----
  int zn = NBT + 2 * N_NODES;
  zero_kernel<<<(zn + 255) / 256, 256, 0, stream>>>(agg, zn);
  degree_kernel<<<(N_EDGES + 255) / 256, 256, 0, stream>>>(src, dst, degO, degI, N_EDGES);
  norm_kernel<<<(N_NODES + 255) / 256, 256, 0, stream>>>(degO, degI, N_NODES);
  edge_scatter<<<N_EDGES, 256, 0, stream>>>(in_feat, src, dst, ew, degO, agg);
  build_x0<<<(MROWS * K0 + 255) / 256, 256, 0, stream>>>(agg, degI, w1, b1, X0);

  // ---- weight precision conversion + bias folding ----
  f2bf_kernel<<<(G4 * K0 + 255) / 256, 256, 0, stream>>>(Wih0, W0b, G4 * K0);
  f2bf_kernel<<<(G4 * HID + 255) / 256, 256, 0, stream>>>(Whh0, Whh0b, G4 * HID);
  f2bf_kernel<<<(G4 * HID + 255) / 256, 256, 0, stream>>>(Wih1, Wih1b, G4 * HID);
  f2bf_kernel<<<(G4 * HID + 255) / 256, 256, 0, stream>>>(Whh1, Whh1b, G4 * HID);
  bias_combine<<<(G4 + 255) / 256, 256, 0, stream>>>(bih0, bhh0, bias0, G4);
  bias_combine<<<(G4 + 255) / 256, 256, 0, stream>>>(bih1, bhh1, bias1, G4);

  // ---- layer 0: xW0 = X0 * Wih0^T + (bih0+bhh0), then recurrence ----
  wmma_gemm_abT<<<dim3(MROWS / 64, G4 / 64), 256, 0, stream>>>(X0, W0b, bias0, xW, MROWS, G4, K0);
  lstm_layer<<<1, 256, 0, stream>>>(xW, Whh0b, hseq0, hlast);

  // ---- layer 1: xW1 = h0 * Wih1^T + (bih1+bhh1), then recurrence ----
  wmma_gemm_abT<<<dim3(MROWS / 64, G4 / 64), 256, 0, stream>>>(hseq0, Wih1b, bias1, xW, MROWS, G4, HID);
  lstm_layer<<<1, 256, 0, stream>>>(xW, Whh1b, hseq1, hlast);

  // ---- FC on last timestep, output [N, B, OUT] ----
  fc_kernel<<<(BATCH * N_NODES * OUTF + 255) / 256, 256, 0, stream>>>(hlast, Wfc, bfc, out);
}
